// ConvKernel_71975061946734
// MI455X (gfx1250) — matrix operand order repaired and checked by reading, not compiled, not run on hardware
//
#include <hip/hip_runtime.h>
#include <hip/hip_bf16.h>
#include <stdint.h>

// ---------------------------------------------------------------------------
// Problem constants (match reference)
// ---------------------------------------------------------------------------
#define NNODES 20000
#define NEDGE  60000
#define WIDTH  256
#define NHEAD  8
#define DHEAD  32
#define BAG    3
#define HOP    3
#define KERNELS 3
#define EPSV   1e-5f

// ---------------------------------------------------------------------------
// WMMA types (CDNA5 / gfx1250, wave32)
// ---------------------------------------------------------------------------
typedef __attribute__((ext_vector_type(16))) __bf16 v16bf;
typedef __attribute__((ext_vector_type(8)))  float  v8f;

// f32 -> bf16 (RNE) bit helpers — used only in producer kernels, never in the
// GEMM hot loop.
__device__ __forceinline__ uint16_t f2bf_bits(float f) {
  union { float f; uint32_t u; } c; c.f = f;
  uint32_t u = c.u + 0x7FFFu + ((c.u >> 16) & 1u);
  return (uint16_t)(u >> 16);
}
__device__ __forceinline__ uint32_t pack_bf16x2(float lo, float hi) {
  union { float f; uint32_t u; } a, b;
  a.f = lo; b.f = hi;
  uint32_t ua = a.u + 0x7FFFu + ((a.u >> 16) & 1u);
  uint32_t ub = b.u + 0x7FFFu + ((b.u >> 16) & 1u);
  return (ua >> 16) | (ub & 0xFFFF0000u);
}

union Frag16 {            // 16 bf16 striped over 8 VGPR words = 2x b128 loads
  v16bf    v;
  uint4    q[2];
  uint32_t u[8];
};

// ---------------------------------------------------------------------------
// GEMM:  Y[m, n] = sum_k X[m, k] * W[n, k]      (Y = X @ W^T)
// X, W are PRE-PACKED bf16; Y is fp32.  v_wmma_f32_16x16x32_bf16 compute.
// One wave owns NT consecutive 16x16 N-tiles: A fragment loaded once per
// K-step.  B tiles are fetched in chunks of <=4 (one clause, one wait), then
// the chunk's WMMAs issue back-to-back (ISA 7.5 multi-WMMA issue pattern).
// blockIdx.y = group g: X += g*gxoff, Y += g*gyoff, W += g*gwoff (elements).
// Requires M%16==0, Ncol%(16*NT)==0, K%32==0 (true for all shapes here).
// ---------------------------------------------------------------------------
template <int NT, bool RELU>
__global__ __launch_bounds__(256)
void gemm_bf16_wmma(float* __restrict__ Y, int ldy,
                    const uint16_t* __restrict__ X, int ldx,
                    const uint16_t* __restrict__ W,
                    int M, int Ncol, int K,
                    long gxoff, long gyoff, long gwoff)
{
  const int g = blockIdx.y;
  X += (long)g * gxoff;
  Y += (long)g * gyoff;
  W += (long)g * gwoff;

  const int lane = threadIdx.x & 31;
  const int wv   = threadIdx.x >> 5;              // 8 waves / block
  const int ngrp = (Ncol >> 4) / NT;              // N tile-groups
  const long tile   = (long)blockIdx.x * 8 + wv;
  const long ntiles = (long)(M >> 4) * ngrp;
  if (tile >= ntiles) return;
  const int tm  = (int)(tile / ngrp);
  const int tnb = (int)(tile % ngrp) * NT;        // first N-tile of this wave

  const int half = lane >> 4;                     // K half-lane split (ISA 7.12.2)
  const int lid  = lane & 15;
  const int arow = tm * 16 + lid;                 // A: M index for this lane

  v8f acc[NT];
#pragma unroll
  for (int t = 0; t < NT; ++t) acc[t] = (v8f){0.f,0.f,0.f,0.f,0.f,0.f,0.f,0.f};

  const uint16_t* xrow = X + (long)arow * ldx;
  constexpr int CH = (NT > 4) ? 4 : NT;           // B-fragment chunk size

  for (int k0 = 0; k0 < K; k0 += 32) {
    // A 16x32 bf16 (ISA 7.12.2): VGPR v<4 -> K = 2v + half*8 ;
    //                            v>=4    -> K = 16 + 2(v-4) + half*8
    const uint16_t* xr = xrow + k0 + half * 8;
    if (k0 + 32 < K) __builtin_prefetch(xr + 32, 0, 1);   // global_prefetch_b8
    Frag16 a;
    a.q[0] = *(const uint4*)(xr);          // bf16 elems 0..7
    a.q[1] = *(const uint4*)(xr + 16);     // bf16 elems 16..23

#pragma unroll
    for (int tc = 0; tc < NT; tc += CH) {
      // chunk of B tiles: issue all loads first (one clause, one wait) ...
      Frag16 b[CH];
#pragma unroll
      for (int t = 0; t < CH; ++t) {
        const int bcol = (tnb + tc + t) * 16 + lid;
        const uint16_t* wr = W + (long)bcol * K + k0 + half * 8;
        b[t].q[0] = *(const uint4*)(wr);
        b[t].q[1] = *(const uint4*)(wr + 16);
      }
      // ... then the chunk's WMMAs back-to-back
#pragma unroll
      for (int t = 0; t < CH; ++t) {
        acc[tc + t] = __builtin_amdgcn_wmma_f32_16x16x32_bf16(
            /*neg_a=*/false, a.v, /*neg_b=*/false, b[t].v,
            /*c_mod=*/(short)0, acc[tc + t],
            /*reuse_a=*/false, /*reuse_b=*/false);
      }
    }
  }

  // D 16x16 f32: lane lid = N, VGPR v -> M = v + 8*half
#pragma unroll
  for (int t = 0; t < NT; ++t) {
#pragma unroll
    for (int v = 0; v < 8; ++v) {
      const int m = tm * 16 + v + 8 * half;
      float val = acc[t][v];
      if (RELU) val = fmaxf(val, 0.f);
      Y[(long)m * ldy + (tnb + t) * 16 + lid] = val;
    }
  }
}

// ---------------------------------------------------------------------------
// f32 -> packed bf16 pair conversion (for weights)
// ---------------------------------------------------------------------------
__global__ void cvt_f32_bf16(uint32_t* __restrict__ o, const float2* __restrict__ in,
                             long npairs) {
  long i = (long)blockIdx.x * blockDim.x + threadIdx.x;
  if (i < npairs) {
    const float2 f = in[i];
    o[i] = pack_bf16x2(f.x, f.y);
  }
}

// ---------------------------------------------------------------------------
// Edge kernel: xx = xs[src] + xt[dst]; GroupNorm(8 heads of 32) via wave32
// shuffles; EmbeddingBag(mean, padding_idx=0) gate bias.  Writes bf16
// operands for the grouped GEMMs directly.
// One block = one edge; wave w = head w (32 lanes == 32 channels).
// ---------------------------------------------------------------------------
__global__ __launch_bounds__(256)
void edge_prep(const float* __restrict__ xs, const float* __restrict__ xt,
               const int* __restrict__ src, const int* __restrict__ dst,
               const int* __restrict__ ea, const float* __restrict__ emb,
               uint16_t* __restrict__ xn_h, uint16_t* __restrict__ gin_h)
{
  const int e    = blockIdx.x;
  const int lane = threadIdx.x & 31;
  const int grp  = threadIdx.x >> 5;
  const int c    = grp * DHEAD + lane;

  const int s = src[e];
  const int d = dst[e];
  float v = xs[(long)s * WIDTH + c] + xt[(long)d * WIDTH + c];

  // per-head mean / biased var over 32 lanes
  float sum = v, sq = v * v;
#pragma unroll
  for (int off = 16; off > 0; off >>= 1) {
    sum += __shfl_xor(sum, off, 32);
    sq  += __shfl_xor(sq,  off, 32);
  }
  const float mu  = sum * (1.f / DHEAD);
  const float var = sq * (1.f / DHEAD) - mu * mu;
  const float xn  = (v - mu) * rsqrtf(var + EPSV);

  // EmbeddingBag mean with padding_idx=0 (row 0 of emb is zeros)
  const int i0 = ea[e * BAG + 0];
  const int i1 = ea[e * BAG + 1];
  const int i2 = ea[e * BAG + 2];
  int cnt = (i0 != 0) + (i1 != 0) + (i2 != 0);
  if (cnt < 1) cnt = 1;
  const float gb = (emb[(long)i0 * WIDTH + c] + emb[(long)i1 * WIDTH + c] +
                    emb[(long)i2 * WIDTH + c]) / (float)cnt;

  xn_h [(long)e * WIDTH + c] = f2bf_bits(xn);
  gin_h[(long)e * WIDTH + c] = f2bf_bits(xn + gb);
}

// ---------------------------------------------------------------------------
// GroupNorm (metaformer): read f32 rows, write bf16 (grouped-GEMM operand)
// ---------------------------------------------------------------------------
__global__ __launch_bounds__(256)
void gnorm_rows(const float* __restrict__ x, uint16_t* __restrict__ o)
{
  const long r   = blockIdx.x;
  const int lane = threadIdx.x & 31;
  const int grp  = threadIdx.x >> 5;
  const long idx = r * WIDTH + grp * DHEAD + lane;
  float v = x[idx];
  float sum = v, sq = v * v;
#pragma unroll
  for (int off = 16; off > 0; off >>= 1) {
    sum += __shfl_xor(sum, off, 32);
    sq  += __shfl_xor(sq,  off, 32);
  }
  const float mu  = sum * (1.f / DHEAD);
  const float var = sq * (1.f / DHEAD) - mu * mu;
  o[idx] = f2bf_bits((v - mu) * rsqrtf(var + EPSV));
}

// ---------------------------------------------------------------------------
// Elementwise helpers
// ---------------------------------------------------------------------------
__global__ void fill_zero(float* __restrict__ p, long n) {
  long i = (long)blockIdx.x * blockDim.x + threadIdx.x;
  if (i < n) p[i] = 0.f;
}
__global__ void copy_f32(float* __restrict__ d, const float* __restrict__ s, long n) {
  long i = (long)blockIdx.x * blockDim.x + threadIdx.x;
  if (i < n) d[i] = s[i];
}
// t = a * b, packed-pair output in bf16 (operand of the post GEMM)
__global__ void ewise_mul_bf16(uint32_t* __restrict__ o, const float2* __restrict__ a,
                               const float2* __restrict__ b, long npairs) {
  long i = (long)blockIdx.x * blockDim.x + threadIdx.x;
  if (i < npairs) {
    const float2 x = a[i], y = b[i];
    o[i] = pack_bf16x2(x.x * y.x, x.y * y.y);
  }
}
__global__ void scatter_add(float* __restrict__ acc, const float* __restrict__ eo,
                            const int* __restrict__ dst, long n) {
  long i = (long)blockIdx.x * blockDim.x + threadIdx.x;
  if (i < n) {
    const int e = (int)(i >> 8);          // WIDTH == 256
    const int c = (int)(i & 255);
    atomicAdd(&acc[(long)dst[e] * WIDTH + c], eo[i]);
  }
}
// x_hop(bf16) = deg^pdeg * acc ; x_out += same (f32)
__global__ void degree_scale_accum(uint16_t* __restrict__ xhop_h,
                                   float* __restrict__ xout,
                                   const float* __restrict__ acc,
                                   const float* __restrict__ deg,
                                   const float* __restrict__ pdeg, long n) {
  long i = (long)blockIdx.x * blockDim.x + threadIdx.x;
  if (i < n) {
    const int nidx = (int)(i >> 8);
    const int c    = (int)(i & 255);
    const float v  = powf(deg[nidx], pdeg[c]) * acc[i];
    xhop_h[i] = f2bf_bits(v);
    xout[i] += v;
  }
}
// xx(f32) = exp(sca_pre)*x_kernel + res ; also bf16 copy (pre-GEMM operand)
__global__ void mf_pre(float* __restrict__ xx, uint16_t* __restrict__ xx_h,
                       const float* __restrict__ xk, const float* __restrict__ res,
                       const float* __restrict__ sca_pre, long n) {
  long i = (long)blockIdx.x * blockDim.x + threadIdx.x;
  if (i < n) {
    const int c = (int)(i & 255);
    const float v = __expf(sca_pre[c]) * xk[i] + res[i];
    xx[i] = v;
    xx_h[i] = f2bf_bits(v);
  }
}
// v = exp(sca_post)*xx + ff ; new x_kernel (f32) and new x_hop (bf16)
__global__ void mf_final(float* __restrict__ xx, float* __restrict__ xk,
                         uint16_t* __restrict__ xhop_h,
                         const float* __restrict__ ff,
                         const float* __restrict__ sca_post, long n) {
  long i = (long)blockIdx.x * blockDim.x + threadIdx.x;
  if (i < n) {
    const int c = (int)(i & 255);
    const float v = __expf(sca_post[c]) * xx[i] + ff[i];
    xx[i] = v;
    xk[i] = v;
    xhop_h[i] = f2bf_bits(v);
  }
}
__global__ void cvt_node_bf16(uint16_t* __restrict__ o, const float* __restrict__ s,
                              long n) {
  long i = (long)blockIdx.x * blockDim.x + threadIdx.x;
  if (i < n) o[i] = f2bf_bits(s[i]);
}
__global__ void final_add(float* __restrict__ out, const float* __restrict__ xout,
                          const float* __restrict__ xres, long n) {
  long i = (long)blockIdx.x * blockDim.x + threadIdx.x;
  if (i < n) out[i] = xout[i] + xres[i];
}

// ---------------------------------------------------------------------------
// Host orchestration
// ---------------------------------------------------------------------------
extern "C" void kernel_launch(void* const* d_in, const int* in_sizes, int n_in,
                              void* d_out, int out_size, void* d_ws, size_t ws_size,
                              hipStream_t stream)
{
  (void)in_sizes; (void)n_in; (void)out_size; (void)ws_size;

  const float* x     = (const float*)d_in[0];
  const float* x_res = (const float*)d_in[1];
  const float* degv[HOP] = {(const float*)d_in[2], (const float*)d_in[3], (const float*)d_in[4]};
  const int*   ei[HOP]   = {(const int*)d_in[5], (const int*)d_in[6], (const int*)d_in[7]};
  const int*   ea[HOP]   = {(const int*)d_in[8], (const int*)d_in[9], (const int*)d_in[10]};

  // params flattened in jax pytree (sorted-key) order:
  //   "mix": 2 x {gate, post, pre, sca_post, sca_pre, value}
  //   "msg": 9 x {deg, emb, gate, post, src, tgt, value}
  struct MixP { const float *gate, *post, *pre, *sca_post, *sca_pre, *value; };
  struct MsgP { const float *deg, *emb, *gate, *post, *src, *tgt, *value; };
  MixP mix[KERNELS - 1];
  MsgP msg[KERNELS * HOP];
  int p = 11;
  for (int b = 0; b < KERNELS - 1; ++b) {
    mix[b].gate     = (const float*)d_in[p++];
    mix[b].post     = (const float*)d_in[p++];
    mix[b].pre      = (const float*)d_in[p++];
    mix[b].sca_post = (const float*)d_in[p++];
    mix[b].sca_pre  = (const float*)d_in[p++];
    mix[b].value    = (const float*)d_in[p++];
  }
  for (int b = 0; b < KERNELS * HOP; ++b) {
    msg[b].deg   = (const float*)d_in[p++];
    msg[b].emb   = (const float*)d_in[p++];
    msg[b].gate  = (const float*)d_in[p++];
    msg[b].post  = (const float*)d_in[p++];
    msg[b].src   = (const float*)d_in[p++];
    msg[b].tgt   = (const float*)d_in[p++];
    msg[b].value = (const float*)d_in[p++];
  }

  // ---- workspace carve -----------------------------------------------------
  char* ws = (char*)d_ws;
  const size_t EBF = (size_t)NEDGE  * WIDTH * sizeof(float);     // edge f32
  const size_t EBH = (size_t)NEDGE  * WIDTH * sizeof(uint16_t);  // edge bf16
  const size_t NBF = (size_t)NNODES * WIDTH * sizeof(float);     // node f32
  const size_t NBH = (size_t)NNODES * WIDTH * sizeof(uint16_t);  // node bf16
  size_t off = 0;
  float*    ebA  = (float*)   (ws + off); off += EBF;  // gate-out / edge-out
  float*    ebB  = (float*)   (ws + off); off += EBF;  // value-out
  uint16_t* ebh0 = (uint16_t*)(ws + off); off += EBH;  // xn  (bf16)
  uint16_t* ebh1 = (uint16_t*)(ws + off); off += EBH;  // gin / t (bf16)
  float*    nb0  = (float*)   (ws + off); off += NBF;  // xx
  float*    nb1  = (float*)   (ws + off); off += NBF;  // x_out accumulator
  float*    nb2  = (float*)   (ws + off); off += NBF;  // x_kernel
  float*    nb3  = (float*)   (ws + off); off += NBF;  // xs / scatter acc / tmp
  float*    nb4  = (float*)   (ws + off); off += NBF;  // xt / tmp
  uint16_t* nbh  = (uint16_t*)(ws + off); off += NBH;  // x_hop_h / xn_h / t_h
  uint16_t* wpool = (uint16_t*)(ws + off);             // bf16 weights

  const long NN = (long)NNODES * WIDTH;
  const long NE = (long)NEDGE  * WIDTH;
  const dim3 blk(256);
  const dim3 gN((unsigned)((NN + 255) / 256));
  const dim3 gE((unsigned)((NE + 255) / 256));

  // ---- convert all weights to bf16 once (4.4 MB, off critical path) -------
  uint16_t* wcur = wpool;
  auto cvt_w = [&](const float* src, long n) -> uint16_t* {
    uint16_t* dst = wcur; wcur += n;
    const long pairs = n / 2;
    cvt_f32_bf16<<<dim3((unsigned)((pairs + 255) / 256)), blk, 0, stream>>>(
        (uint32_t*)dst, (const float2*)src, pairs);
    return dst;
  };
  struct MixH { uint16_t *gate, *post, *pre, *value; };
  struct MsgH { uint16_t *gate, *post, *src, *tgt, *value; };
  MixH mixh[KERNELS - 1];
  MsgH msgh[KERNELS * HOP];
  const long WD = (long)WIDTH * WIDTH;          // dense weight elems
  const long WG = (long)NHEAD * DHEAD * DHEAD;  // grouped weight elems
  for (int b = 0; b < KERNELS - 1; ++b) {
    mixh[b].gate  = cvt_w(mix[b].gate,  WG);
    mixh[b].post  = cvt_w(mix[b].post,  WD);
    mixh[b].pre   = cvt_w(mix[b].pre,   WD);
    mixh[b].value = cvt_w(mix[b].value, WG);
  }
  for (int b = 0; b < KERNELS * HOP; ++b) {
    msgh[b].gate  = cvt_w(msg[b].gate,  WG);
    msgh[b].post  = cvt_w(msg[b].post,  WD);
    msgh[b].src   = cvt_w(msg[b].src,   WD);
    msgh[b].tgt   = cvt_w(msg[b].tgt,   WD);
    msgh[b].value = cvt_w(msg[b].value, WG);
  }

  // Dense GEMM: Ncol=256, NT=8 -> 2 tile-groups along N.
  auto gemm_dense = [&](float* Y, const uint16_t* X, const uint16_t* W, int M) {
    const long tiles = (long)(M / 16) * ((WIDTH / 16) / 8);
    dim3 grid((unsigned)((tiles + 7) / 8), 1);
    gemm_bf16_wmma<8, false><<<grid, blk, 0, stream>>>(
        Y, WIDTH, X, WIDTH, W, M, WIDTH, WIDTH, 0, 0, 0);
  };
  // Grouped 1x1 conv: 8 groups of 32x32, NT=2 covers the whole group width.
  auto gemm_group = [&](float* Y, const uint16_t* X, const uint16_t* W, int M,
                        bool relu) {
    const long tiles = (long)(M / 16);
    dim3 grid((unsigned)((tiles + 7) / 8), NHEAD);
    if (relu)
      gemm_bf16_wmma<2, true><<<grid, blk, 0, stream>>>(
          Y, WIDTH, X, WIDTH, W, M, DHEAD, DHEAD,
          DHEAD, DHEAD, (long)DHEAD * DHEAD);
    else
      gemm_bf16_wmma<2, false><<<grid, blk, 0, stream>>>(
          Y, WIDTH, X, WIDTH, W, M, DHEAD, DHEAD,
          DHEAD, DHEAD, (long)DHEAD * DHEAD);
  };

  // ---- init: x_hop(bf16) = x ; x_kernel = x ; x_out = 0 -------------------
  cvt_node_bf16<<<gN, blk, 0, stream>>>(nbh, x, NN);
  copy_f32     <<<gN, blk, 0, stream>>>(nb2, x, NN);
  fill_zero    <<<gN, blk, 0, stream>>>(nb1, NN);

  for (int k = 0; k < KERNELS; ++k) {
    for (int h = 0; h < HOP; ++h) {
      const MsgP& m  = msg[k * HOP + h];
      const MsgH& mh = msgh[k * HOP + h];
      // xs = x @ src^T ; xt = x @ tgt^T   (bf16 in, f32 out)
      gemm_dense(nb3, nbh, mh.src, NNODES);
      gemm_dense(nb4, nbh, mh.tgt, NNODES);
      // edge gather + GroupNorm + EmbeddingBag -> bf16 operands
      edge_prep<<<NEDGE, blk, 0, stream>>>(nb3, nb4, ei[h], ei[h] + NEDGE,
                                           ea[h], m.emb, ebh0, ebh1);
      // grouped 1x1 convs (8 heads of 32x32), relu fused into gate
      gemm_group(ebA, ebh1, mh.gate,  NEDGE, true);
      gemm_group(ebB, ebh0, mh.value, NEDGE, false);
      ewise_mul_bf16<<<dim3((unsigned)(NE / 2 / 256)), blk, 0, stream>>>(
          (uint32_t*)ebh1, (const float2*)ebA, (const float2*)ebB, NE / 2);
      // edge out = t @ post^T
      gemm_dense(ebA, ebh1, mh.post, NEDGE);
      // scatter-sum into nodes, then degree scaling + x_out accumulation
      fill_zero  <<<gN, blk, 0, stream>>>(nb3, NN);
      scatter_add<<<gE, blk, 0, stream>>>(nb3, ebA, ei[h] + NEDGE, NE);
      degree_scale_accum<<<gN, blk, 0, stream>>>(nbh, nb1, nb3, degv[h], m.deg, NN);
    }
    if (k < KERNELS - 1) {
      const MixP& m  = mix[k];
      const MixH& mh = mixh[k];
      mf_pre<<<gN, blk, 0, stream>>>(nb0, nbh, nb2, nb1, m.sca_pre, NN);
      gemm_dense(nb3, nbh, mh.pre, NNODES);
      gnorm_rows<<<NNODES, blk, 0, stream>>>(nb3, nbh);
      gemm_group(nb4, nbh, mh.gate,  NNODES, true);
      gemm_group(nb1, nbh, mh.value, NNODES, false);
      ewise_mul_bf16<<<dim3((unsigned)(NN / 2 / 256)), blk, 0, stream>>>(
          (uint32_t*)nbh, (const float2*)nb4, (const float2*)nb1, NN / 2);
      gemm_dense(nb3, nbh, mh.post, NNODES);
      mf_final<<<gN, blk, 0, stream>>>(nb0, nb2, nbh, nb3, m.sca_post, NN);
      fill_zero<<<gN, blk, 0, stream>>>(nb1, NN);  // x_out = 0
    }
  }

  final_add<<<gN, blk, 0, stream>>>((float*)d_out, nb1, x_res, NN);
}
